// GatedRefAltMLPBlock_54966991454316
// MI455X (gfx1250) — matrix operand order repaired and checked by reading, not compiled, not run on hardware
//
#include <hip/hip_runtime.h>
#include <hip/hip_bf16.h>

#define NREF 131072
#define NALT 32768
#define BSEG 4096
#define DDIM 128
#define FDIM 512
#define HDIM 256

typedef __attribute__((ext_vector_type(16))) __bf16 v16bf;
typedef __attribute__((ext_vector_type(8)))  float  v8f;

union Frag { v16bf v; unsigned int u[8]; };

__device__ __forceinline__ unsigned short f2bf(float f) {
    unsigned int u = __float_as_uint(f);
    u += 0x7FFFu + ((u >> 16) & 1u);            // round-to-nearest-even
    return (unsigned short)(u >> 16);
}
__device__ __forceinline__ float bf2f(unsigned short h) {
    return __uint_as_float(((unsigned int)h) << 16);
}
__device__ __forceinline__ float selu_f(float x) {
    const float sc = 1.0507009873554805f, al = 1.6732632423543772f;
    return x > 0.f ? sc * x : sc * al * (__expf(x) - 1.f);
}

// ---------------------------------------------------------------- prep kernels
__global__ void zero_kernel(float* p, int n) {
    int i = blockIdx.x * blockDim.x + threadIdx.x;
    if (i < n) p[i] = 0.f;
}

// w is K x N (row-major).  wT[n*K + k] = bf16(w[k*N + n])  (N-major, K contiguous)
__global__ void convertT_kernel(const float* __restrict__ w,
                                unsigned short* __restrict__ wT, int K, int N) {
    int i = blockIdx.x * blockDim.x + threadIdx.x;
    if (i >= K * N) return;
    int n = i / K, k = i - n * K;
    wT[i] = f2bf(w[(size_t)k * N + n]);
}

// ----------------------------------------------------- fused LN1 + GEMM1 stage
// block = 128 threads (4 waves), computes 32 rows x 512 cols
// wave w covers columns [w*128, w*128+128) for ALL 32 rows (B reused 2x)
__global__ __launch_bounds__(128) void gemm1_kernel(
    const float* __restrict__ x, const int* __restrict__ seg,
    const float* __restrict__ n1w, const float* __restrict__ n1b,
    const unsigned short* __restrict__ w1T,   // [F][D] bf16 (N-major)
    const float* __restrict__ b1,
    const float* __restrict__ n2w, const float* __restrict__ n2b,
    unsigned short* __restrict__ z1o,         // [rows][H] bf16
    unsigned short* __restrict__ z2o,         // [rows][H] bf16 (post-LN2)
    float* __restrict__ seg_sum,              // [B][H]
    float* __restrict__ seg_cnt,              // [B]
    int nrows)
{
    __shared__ unsigned short Abuf[32][DDIM + 8];
    __shared__ float zbuf[32][FDIM];
    __shared__ float red[32][4][2];
    __shared__ float rowstat[32][2];
    __shared__ int   segs[32];

    const int tid  = threadIdx.x;
    const int row0 = blockIdx.x * 32;
    if (row0 >= nrows) return;

    if (tid < 32) segs[tid] = seg[row0 + tid];

    // ---- LayerNorm over D=128, write bf16 A tile to LDS
    const int r = tid >> 2, p = tid & 3;      // 4 threads per row, 32 cols each
    float xv[32];
    {
        float s = 0.f, ss = 0.f;
#pragma unroll
        for (int i = 0; i < 32; ++i) {
            float v = x[(size_t)(row0 + r) * DDIM + p * 32 + i];
            xv[i] = v; s += v; ss += v * v;
        }
        red[r][p][0] = s; red[r][p][1] = ss;
    }
    __syncthreads();
    if (p == 0) {
        float s0 = 0.f, ss0 = 0.f;
        for (int q = 0; q < 4; ++q) { s0 += red[r][q][0]; ss0 += red[r][q][1]; }
        float m = s0 * (1.f / DDIM);
        float v = ss0 * (1.f / DDIM) - m * m;
        rowstat[r][0] = m; rowstat[r][1] = rsqrtf(v + 1e-5f);
    }
    __syncthreads();
    {
        float m = rowstat[r][0], rs = rowstat[r][1];
#pragma unroll
        for (int i = 0; i < 32; ++i) {
            int c = p * 32 + i;
            Abuf[r][c] = f2bf((xv[i] - m) * rs * n1w[c] + n1b[c]);
        }
    }
    __syncthreads();

    // ---- GEMM1 (B fragments reused for both 16-row groups)
    const int wave = tid >> 5, lane = tid & 31;
    const int lm = lane & 15, hi = lane >> 4;
    v8f zero = {};
    v8f acc[2][8];
#pragma unroll
    for (int g = 0; g < 2; ++g)
#pragma unroll
        for (int nt = 0; nt < 8; ++nt) acc[g][nt] = zero;

#pragma unroll
    for (int kt = 0; kt < 4; ++kt) {
        Frag af[2];
#pragma unroll
        for (int g = 0; g < 2; ++g)
#pragma unroll
            for (int j = 0; j < 8; ++j) {
                int k = kt * 32 + ((j >> 2) << 4) + hi * 8 + ((j & 3) << 1);
                af[g].u[j] = *(const unsigned int*)&Abuf[g * 16 + lm][k];
            }
#pragma unroll
        for (int nt = 0; nt < 8; ++nt) {
            int cg = wave * 128 + nt * 16 + lm;
            Frag bf;
#pragma unroll
            for (int i = 0; i < 8; ++i) {
                int k = kt * 32 + ((i >> 2) << 4) + hi * 8 + ((i & 3) << 1);
                bf.u[i] = *(const unsigned int*)&w1T[(size_t)cg * DDIM + k];
            }
            acc[0][nt] = __builtin_amdgcn_wmma_f32_16x16x32_bf16(
                false, af[0].v, false, bf.v, (short)0, acc[0][nt], false, false);
            acc[1][nt] = __builtin_amdgcn_wmma_f32_16x16x32_bf16(
                false, af[1].v, false, bf.v, (short)0, acc[1][nt], false, false);
        }
    }
    // bias + SELU -> zbuf
#pragma unroll
    for (int nt = 0; nt < 8; ++nt) {
        int cw = wave * 128 + nt * 16 + lm;
        float bb = b1[cw];
#pragma unroll
        for (int g = 0; g < 2; ++g)
#pragma unroll
            for (int rr = 0; rr < 8; ++rr)
                zbuf[g * 16 + hi * 8 + rr][cw] = selu_f(acc[g][nt][rr] + bb);
    }
    __syncthreads();

    // ---- LN2 over cols [256,512)
    {
        float s2 = 0.f, ss2 = 0.f;
        for (int i = 0; i < 64; ++i) {
            float v = zbuf[r][HDIM + p * 64 + i];
            s2 += v; ss2 += v * v;
        }
        red[r][p][0] = s2; red[r][p][1] = ss2;
    }
    __syncthreads();
    if (p == 0) {
        float s0 = 0.f, ss0 = 0.f;
        for (int q = 0; q < 4; ++q) { s0 += red[r][q][0]; ss0 += red[r][q][1]; }
        float m = s0 * (1.f / HDIM);
        float v = ss0 * (1.f / HDIM) - m * m;
        rowstat[r][0] = m; rowstat[r][1] = rsqrtf(v + 1e-5f);
    }
    __syncthreads();

    // ---- write z1 / z2ln (bf16), keep z2ln in LDS for segment sums
    for (int e = 0; e < 64; ++e) {
        int idx = tid * 64 + e;
        int rr = idx >> 8, c = idx & 255;
        z1o[(size_t)(row0 + rr) * HDIM + c] = f2bf(zbuf[rr][c]);
        float m = rowstat[rr][0], rs = rowstat[rr][1];
        float z2ln = (zbuf[rr][HDIM + c] - m) * rs * n2w[c] + n2b[c];
        z2o[(size_t)(row0 + rr) * HDIM + c] = f2bf(z2ln);
        zbuf[rr][HDIM + c] = z2ln;
    }
    __syncthreads();

    // ---- segment sums (segments are sorted -> usually uniform per block)
    if (segs[0] == segs[31]) {
        int sg = segs[0];
#pragma unroll
        for (int cc = 0; cc < 2; ++cc) {
            int c = tid * 2 + cc;
            float sum = 0.f;
#pragma unroll
            for (int rr = 0; rr < 32; ++rr) sum += zbuf[rr][HDIM + c];
            atomicAdd(&seg_sum[(size_t)sg * HDIM + c], sum);
        }
        if (tid == 0) atomicAdd(&seg_cnt[sg], 32.f);
    } else {
        for (int e = 0; e < 64; ++e) {
            int idx = tid * 64 + e;
            int rr = idx >> 8, c = idx & 255;
            atomicAdd(&seg_sum[(size_t)segs[rr] * HDIM + c], zbuf[rr][HDIM + c]);
        }
        if (tid < 32) atomicAdd(&seg_cnt[segs[tid]], 1.f);
    }
}

// ------------------------------------------------- per-segment gate vectors
__global__ void mf_kernel(const float* __restrict__ ref_sum,
                          const float* __restrict__ alt_sum,
                          const float* __restrict__ cnt_r,
                          const float* __restrict__ cnt_a,
                          const float* __restrict__ ref_reg,
                          const float* reg_w_pre, const float* beta_ref,
                          const float* beta_alt, const float* gamma,
                          float* __restrict__ ref_gvec,
                          float* __restrict__ alt_gvec)
{
    int i = blockIdx.x * 256 + threadIdx.x;
    int b = i >> 8, h = i & 255;
    float rw  = __expf(*reg_w_pre) + 0.25f;
    float rmf = (ref_sum[i] + rw * ref_reg[h]) / (cnt_r[b] + rw);
    float amf = alt_sum[i] / fmaxf(cnt_a[b], 1.f);
    ref_gvec[i] = (*beta_ref) * rmf;
    alt_gvec[i] = (*beta_alt) * amf + (*gamma) * rmf;
}

// ------------------------------------------ fused gate + GEMM2 + residual
// block = 128 threads (4 waves), 32 rows x 128 cols; wave = 32 rows x 32 cols
__global__ __launch_bounds__(128) void gemm2_kernel(
    const float* __restrict__ x, const int* __restrict__ seg,
    const unsigned short* __restrict__ z1, const unsigned short* __restrict__ z2,
    const float* __restrict__ gvec, const float* alpha_p,
    const unsigned short* __restrict__ w2T,   // [D][H] bf16 (N-major)
    const float* __restrict__ b2,
    float* __restrict__ out, int nrows)
{
    __shared__ unsigned short Abuf[32][HDIM + 8];
    __shared__ int segs[32];
    const int tid  = threadIdx.x;
    const int row0 = blockIdx.x * 32;
    if (row0 >= nrows) return;
    if (tid < 32) segs[tid] = seg[row0 + tid];
    __syncthreads();
    const float alpha = *alpha_p;

    for (int e = 0; e < 64; ++e) {
        int idx = tid * 64 + e;
        int rr = idx >> 8, c = idx & 255;
        float z1f  = bf2f(z1[(size_t)(row0 + rr) * HDIM + c]);
        float z2f  = bf2f(z2[(size_t)(row0 + rr) * HDIM + c]);
        float gate = z2f * alpha + 1.f + gvec[(size_t)segs[rr] * HDIM + c];
        Abuf[rr][c] = f2bf(z1f * gate);
    }
    __syncthreads();

    const int wave = tid >> 5, lane = tid & 31;
    const int lm = lane & 15, hi = lane >> 4;
    v8f zero = {};
    v8f acc[2][2];
#pragma unroll
    for (int g = 0; g < 2; ++g) { acc[g][0] = zero; acc[g][1] = zero; }

#pragma unroll
    for (int kt = 0; kt < 8; ++kt) {
        Frag af[2];
#pragma unroll
        for (int g = 0; g < 2; ++g)
#pragma unroll
            for (int j = 0; j < 8; ++j) {
                int k = kt * 32 + ((j >> 2) << 4) + hi * 8 + ((j & 3) << 1);
                af[g].u[j] = *(const unsigned int*)&Abuf[g * 16 + lm][k];
            }
#pragma unroll
        for (int nt = 0; nt < 2; ++nt) {
            int cg = wave * 32 + nt * 16 + lm;
            Frag bf;
#pragma unroll
            for (int i = 0; i < 8; ++i) {
                int k = kt * 32 + ((i >> 2) << 4) + hi * 8 + ((i & 3) << 1);
                bf.u[i] = *(const unsigned int*)&w2T[(size_t)cg * HDIM + k];
            }
            acc[0][nt] = __builtin_amdgcn_wmma_f32_16x16x32_bf16(
                false, af[0].v, false, bf.v, (short)0, acc[0][nt], false, false);
            acc[1][nt] = __builtin_amdgcn_wmma_f32_16x16x32_bf16(
                false, af[1].v, false, bf.v, (short)0, acc[1][nt], false, false);
        }
    }
#pragma unroll
    for (int nt = 0; nt < 2; ++nt) {
        int cw = wave * 32 + nt * 16 + lm;
        float bb = b2[cw];
#pragma unroll
        for (int g = 0; g < 2; ++g)
#pragma unroll
            for (int rr = 0; rr < 8; ++rr) {
                size_t gi = (size_t)(row0 + g * 16 + hi * 8 + rr) * DDIM + cw;
                out[gi] = x[gi] + acc[g][nt][rr] + bb;
            }
    }
}

// --------------------------------------------------------------------- launch
extern "C" void kernel_launch(void* const* d_in, const int* in_sizes, int n_in,
                              void* d_out, int out_size, void* d_ws, size_t ws_size,
                              hipStream_t stream)
{
    const float* ref_flat  = (const float*)d_in[0];
    const float* alt_flat  = (const float*)d_in[1];
    const int*   ref_seg   = (const int*)d_in[2];
    const int*   alt_seg   = (const int*)d_in[3];
    const float* n1w       = (const float*)d_in[4];
    const float* n1b       = (const float*)d_in[5];
    const float* w1_ref    = (const float*)d_in[6];
    const float* b1_ref    = (const float*)d_in[7];
    const float* w1_alt    = (const float*)d_in[8];
    const float* b1_alt    = (const float*)d_in[9];
    const float* n2w       = (const float*)d_in[10];
    const float* n2b       = (const float*)d_in[11];
    const float* alpha_ref = (const float*)d_in[12];
    const float* alpha_alt = (const float*)d_in[13];
    const float* beta_ref  = (const float*)d_in[14];
    const float* beta_alt  = (const float*)d_in[15];
    const float* gamma     = (const float*)d_in[16];
    const float* ref_reg   = (const float*)d_in[17];
    const float* reg_w_pre = (const float*)d_in[18];
    const float* w2_ref    = (const float*)d_in[19];
    const float* b2_ref    = (const float*)d_in[20];
    const float* w2_alt    = (const float*)d_in[21];
    const float* b2_alt    = (const float*)d_in[22];

    char* ws = (char*)d_ws;
    size_t off = 0;
    auto alloc = [&](size_t bytes) -> char* {
        char* r = ws + off;
        off += (bytes + 255) & ~(size_t)255;
        return r;
    };
    unsigned short* w1rT = (unsigned short*)alloc((size_t)FDIM * DDIM * 2);
    unsigned short* w1aT = (unsigned short*)alloc((size_t)FDIM * DDIM * 2);
    unsigned short* w2rT = (unsigned short*)alloc((size_t)DDIM * HDIM * 2);
    unsigned short* w2aT = (unsigned short*)alloc((size_t)DDIM * HDIM * 2);
    float* ref_sum = (float*)alloc((size_t)BSEG * HDIM * 4);   // contiguous zero region start
    float* alt_sum = (float*)alloc((size_t)BSEG * HDIM * 4);
    float* cnt_r   = (float*)alloc((size_t)BSEG * 4);
    float* cnt_a   = (float*)alloc((size_t)BSEG * 4);          // contiguous zero region end
    float* ref_gv  = (float*)alloc((size_t)BSEG * HDIM * 4);
    float* alt_gv  = (float*)alloc((size_t)BSEG * HDIM * 4);
    unsigned short* z1r = (unsigned short*)alloc((size_t)NREF * HDIM * 2);
    unsigned short* z2r = (unsigned short*)alloc((size_t)NREF * HDIM * 2);
    unsigned short* z1a = (unsigned short*)alloc((size_t)NALT * HDIM * 2);
    unsigned short* z2a = (unsigned short*)alloc((size_t)NALT * HDIM * 2);

    // prep: zero accumulators (every call) + bf16-transpose weights
    {
        int nz = 2 * BSEG * HDIM + 2 * BSEG;
        zero_kernel<<<(nz + 255) / 256, 256, 0, stream>>>(ref_sum, nz);
        int n1 = DDIM * FDIM;
        convertT_kernel<<<(n1 + 255) / 256, 256, 0, stream>>>(w1_ref, w1rT, DDIM, FDIM);
        convertT_kernel<<<(n1 + 255) / 256, 256, 0, stream>>>(w1_alt, w1aT, DDIM, FDIM);
        int n2 = HDIM * DDIM;
        convertT_kernel<<<(n2 + 255) / 256, 256, 0, stream>>>(w2_ref, w2rT, HDIM, DDIM);
        convertT_kernel<<<(n2 + 255) / 256, 256, 0, stream>>>(w2_alt, w2aT, HDIM, DDIM);
    }

    // stage 1: LN1 + GEMM1 + SELU + LN2 + segment accumulation
    gemm1_kernel<<<NREF / 32, 128, 0, stream>>>(ref_flat, ref_seg, n1w, n1b, w1rT, b1_ref,
                                                n2w, n2b, z1r, z2r, ref_sum, cnt_r, NREF);
    gemm1_kernel<<<NALT / 32, 128, 0, stream>>>(alt_flat, alt_seg, n1w, n1b, w1aT, b1_alt,
                                                n2w, n2b, z1a, z2a, alt_sum, cnt_a, NALT);

    // stage 2: segment-mean gate vectors
    mf_kernel<<<BSEG * HDIM / 256, 256, 0, stream>>>(ref_sum, alt_sum, cnt_r, cnt_a, ref_reg,
                                                     reg_w_pre, beta_ref, beta_alt, gamma,
                                                     ref_gv, alt_gv);

    // stage 3: gate + GEMM2 + residual
    float* out_r = (float*)d_out;
    float* out_a = out_r + (size_t)NREF * DDIM;
    gemm2_kernel<<<NREF / 32, 128, 0, stream>>>(ref_flat, ref_seg, z1r, z2r, ref_gv,
                                                alpha_ref, w2rT, b2_ref, out_r, NREF);
    gemm2_kernel<<<NALT / 32, 128, 0, stream>>>(alt_flat, alt_seg, z1a, z2a, alt_gv,
                                                alpha_alt, w2aT, b2_alt, out_a, NALT);
}
